// BirdJEPA_75118978007608
// MI455X (gfx1250) — matrix operand order repaired and checked by reading, compiled, not run on hardware
//
#include <hip/hip_runtime.h>
#include <stdint.h>

#define BB 4
#define TT 2048
#define FIN0 513
#define HID 256
#define HEADS 8
#define HD 32
#define MLPD 1024

typedef _Float16 v16h __attribute__((ext_vector_type(16)));
typedef float v8f __attribute__((ext_vector_type(8)));

#define WMMA_F16(a, b, c) __builtin_amdgcn_wmma_f32_16x16x32_f16(false, (a), false, (b), (short)0, (c), false, false)

__device__ __forceinline__ float gelu_exact(float x) {
    return 0.5f * x * (1.0f + erff(x * 0.70710678118654752440f));
}

// ---------------------------------------------------------------------------
// Weight packing into the contiguous-k WMMA A-operand order used here (16x32 f16 per tile-chunk):
//   lane<16 : M=lane,   elem e -> K = e                           ({0..15})
//   lane>=16: M=lane-16,elem e -> K = e + 16                      ({16..31})
// ---------------------------------------------------------------------------

// dst layout: [tap = kh*KW+kw][cot][kc][lane(32)][e(16)]
__global__ void pack_conv_w(const float* __restrict__ w, _Float16* __restrict__ dst,
                            int COUT, int CIN, int KH, int KW) {
    long total = (long)KH * KW * (COUT >> 4) * (CIN >> 5) * 512;
    long i = blockIdx.x * (long)blockDim.x + threadIdx.x;
    if (i >= total) return;
    int e = (int)(i & 15); long r = i >> 4;
    int lane = (int)(r & 31); r >>= 5;
    int kcn = CIN >> 5;
    int kc = (int)(r % kcn); r /= kcn;
    int cotn = COUT >> 4;
    int cot = (int)(r % cotn); r /= cotn;
    int tap = (int)r;
    int kh = tap / KW, kw = tap % KW;
    int m = cot * 16 + (lane & 15);
    int koff = e + ((lane >= 16) ? 16 : 0);
    int ci = kc * 32 + koff;
    dst[i] = (_Float16)w[(((long)m * CIN + ci) * KH + kh) * KW + kw];
}

// dst layout: [mt][kc][lane][e]; perm==1: k = f*128+c maps to src col c*33+f (proj flatten)
__global__ void pack_mat_w(const float* __restrict__ w, _Float16* __restrict__ dst,
                           int M, int K, int perm) {
    long total = (long)M * K;
    long i = blockIdx.x * (long)blockDim.x + threadIdx.x;
    if (i >= total) return;
    int e = (int)(i & 15); long r = i >> 4;
    int lane = (int)(r & 31); r >>= 5;
    int kcn = K >> 5;
    int kc = (int)(r % kcn); r /= kcn;
    int mt = (int)r;
    int m = mt * 16 + (lane & 15);
    int k = kc * 32 + e + ((lane >= 16) ? 16 : 0);
    int ks = k;
    if (perm) { int f = k >> 7, c = k & 127; ks = c * 33 + f; }
    dst[i] = (_Float16)w[(long)m * K + ks];
}

// ---------------------------------------------------------------------------
// conv1: cin=1, k=3, stride(2,1), pad 1 -> VALU direct; fused BN + GELU.
// out layout channels-last f16: [B][257][T][32]
// ---------------------------------------------------------------------------
__global__ void conv1_kernel(const float* __restrict__ x, const float* __restrict__ w,
                             const float* __restrict__ bias, const float* __restrict__ bns,
                             const float* __restrict__ bnb, const float* __restrict__ bnm,
                             const float* __restrict__ bnv, _Float16* __restrict__ out) {
    long i = blockIdx.x * (long)blockDim.x + threadIdx.x;
    long total = (long)BB * 257 * TT * 32;
    if (i >= total) return;
    int co = (int)(i & 31); long r = i >> 5;
    int t = (int)(r % TT); r /= TT;
    int fo = (int)(r % 257);
    int b = (int)(r / 257);
    float acc = 0.0f;
    for (int kh = 0; kh < 3; kh++) {
        int fi = 2 * fo + kh - 1;
        if (fi < 0 || fi >= FIN0) continue;
        for (int kw = 0; kw < 3; kw++) {
            int ti = t + kw - 1;
            if (ti < 0 || ti >= TT) continue;
            acc += w[co * 9 + kh * 3 + kw] * x[((long)b * FIN0 + fi) * TT + ti];
        }
    }
    float s = bns[co] * rsqrtf(bnv[co] + 1e-5f);
    float y = (acc + bias[co] - bnm[co]) * s + bnb[co];
    out[i] = (_Float16)gelu_exact(y);
}

// ---------------------------------------------------------------------------
// Generic implicit-GEMM conv via per-tap WMMA. One wave -> 16(co) x 16(t) tile.
// in : channels-last f16 [B][FIN][T][CIN]; out: [B][FOUT][T][COUT]
// ---------------------------------------------------------------------------
__global__ void conv_wmma(const _Float16* __restrict__ in, const _Float16* __restrict__ wp,
                          const float* __restrict__ bias, const float* __restrict__ bns,
                          const float* __restrict__ bnb, const float* __restrict__ bnm,
                          const float* __restrict__ bnv, _Float16* __restrict__ out,
                          int CIN, int COUT, int FIN, int FOUT, int KS, int PAD) {
    int lane = threadIdx.x;
    int t0 = blockIdx.x * 16;
    int fo = blockIdx.y;
    int cotn = COUT >> 4;
    int b = blockIdx.z / cotn;
    int cot = blockIdx.z % cotn;
    int kcn = CIN >> 5;
    int nlo = lane & 15, hi = lane >> 4;
    v8f c = {};
    for (int kh = 0; kh < KS; kh++) {
        int fi = 2 * fo + kh - PAD;
        if (fi < 0 || fi >= FIN) continue;
        const _Float16* inrow = in + ((long)b * FIN + fi) * TT * CIN;
        for (int kw = 0; kw < KS; kw++) {
            int t = t0 + nlo + kw - PAD;
            bool ok = (t >= 0 && t < TT);
            for (int kc = 0; kc < kcn; kc++) {
                v16h a = *(const v16h*)(wp + ((((long)(kh * KS + kw) * cotn + cot) * kcn + kc) * 32 + lane) * 16);
                v16h bm = {};
                if (ok) bm = *(const v16h*)(inrow + (long)t * CIN + kc * 32 + hi * 16);
                c = WMMA_F16(a, bm, c);
            }
        }
    }
    int t = t0 + nlo;
    _Float16* orow = out + (((long)b * FOUT + fo) * TT + t) * COUT + cot * 16 + hi * 8;
    for (int r2 = 0; r2 < 8; r2++) {
        int co = cot * 16 + hi * 8 + r2;
        float s = bns[co] * rsqrtf(bnv[co] + 1e-5f);
        float y = (c[r2] + bias[co] - bnm[co]) * s + bnb[co];
        orow[r2] = (_Float16)gelu_exact(y);
    }
}

// ---------------------------------------------------------------------------
// Generic WMMA GEMM: D[n,m] = sum_k A[m,k]*B[n,k]  (A pre-packed, B row-major f16)
// ---------------------------------------------------------------------------
#define GM_PROJ 0  // f32 out = acc + bias[m] + sine_pos_enc(t, m); B from conv4 layout
#define GM_QKV  1  // f32 out = acc + bias[m]
#define GM_RES  2  // f32 out = acc + bias[m] + res[n,m]
#define GM_GELU 3  // f16 out = gelu(acc + bias[m])

__global__ void gemm_wmma(const _Float16* __restrict__ ap, const _Float16* __restrict__ bsrc,
                          int M, int K, int mode,
                          const float* __restrict__ bias, const float* __restrict__ res,
                          float* __restrict__ outf, _Float16* __restrict__ outh) {
    int lane = threadIdx.x;
    int n0 = blockIdx.x * 16;
    int mt = blockIdx.y;
    int kcn = K >> 5;
    int nlo = lane & 15, hi = lane >> 4;
    int n = n0 + nlo;
    v8f c = {};
    const _Float16* arow = ap + (long)mt * kcn * 512 + lane * 16;
    if (mode == GM_PROJ) {
        int b = n / TT, t = n % TT;
        for (int kc = 0; kc < kcn; kc++) {
            int k0 = kc * 32;
            int f = k0 >> 7, cbase = k0 & 127;
            v16h a = *(const v16h*)(arow + (long)kc * 512);
            v16h bm = *(const v16h*)(bsrc + (((long)b * 33 + f) * TT + t) * 128 + cbase + hi * 16);
            c = WMMA_F16(a, bm, c);
        }
    } else {
        const _Float16* brow = bsrc + (long)n * K + hi * 16;
        for (int kc = 0; kc < kcn; kc++) {
            v16h a = *(const v16h*)(arow + (long)kc * 512);
            v16h bm = *(const v16h*)(brow + kc * 32);
            c = WMMA_F16(a, bm, c);
        }
    }
    for (int r2 = 0; r2 < 8; r2++) {
        int m = mt * 16 + hi * 8 + r2;
        float v = c[r2] + bias[m];
        if (mode == GM_PROJ) {
            int t = n % TT;
            int pair = m >> 1;
            float div = expf(-9.210340371976184f * (float)(2 * pair) / (float)HID);
            float ang = (float)t * div;
            v += (m & 1) ? cosf(ang) : sinf(ang);
            outf[(long)n * M + m] = v;
        } else if (mode == GM_QKV) {
            outf[(long)n * M + m] = v;
        } else if (mode == GM_RES) {
            outf[(long)n * M + m] = v + res[(long)n * M + m];
        } else {
            outh[(long)n * M + m] = (_Float16)gelu_exact(v);
        }
    }
}

// ---------------------------------------------------------------------------
// LayerNorm: one wave32 per row of 256; f32 in -> f16 out.
// ---------------------------------------------------------------------------
__global__ void layernorm_k(const float* __restrict__ x, const float* __restrict__ s,
                            const float* __restrict__ bsk, _Float16* __restrict__ out) {
    int row = blockIdx.x;
    int lane = threadIdx.x;
    const float* xr = x + (long)row * HID;
    float v[8];
    float sum = 0.0f;
    for (int j = 0; j < 8; j++) { v[j] = xr[lane + 32 * j]; sum += v[j]; }
    for (int o = 16; o >= 1; o >>= 1) sum += __shfl_xor(sum, o, 32);
    float mu = sum * (1.0f / HID);
    float vs = 0.0f;
    for (int j = 0; j < 8; j++) { float d = v[j] - mu; vs += d * d; }
    for (int o = 16; o >= 1; o >>= 1) vs += __shfl_xor(vs, o, 32);
    float inv = rsqrtf(vs * (1.0f / HID) + 1e-5f);
    for (int j = 0; j < 8; j++) {
        int m = lane + 32 * j;
        out[(long)row * HID + m] = (_Float16)((v[j] - mu) * inv * s[m] + bsk[m]);
    }
}

// ---------------------------------------------------------------------------
// Local attention: window |q-k|<=8 -> <=17 keys per query. Thread per (b,h,t).
// qkv f32 layout [(b*T+t)*768 + {0,256,512} + h*32 + d]; o f16 [(b*T+t)*256 + h*32 + d]
// ---------------------------------------------------------------------------
__global__ void attn_local(const float* __restrict__ qkv, _Float16* __restrict__ o) {
    long i = blockIdx.x * (long)blockDim.x + threadIdx.x;
    long total = (long)BB * HEADS * TT;
    if (i >= total) return;
    int t = (int)(i % TT); long r = i / TT;
    int h = (int)(r % HEADS);
    int b = (int)(r / HEADS);
    const float scale = 0.17677669529663687f; // 1/sqrt(32)
    float qr[HD];
    const float* qp = qkv + ((long)b * TT + t) * 768 + h * HD;
    for (int d = 0; d < HD; d++) qr[d] = qp[d];
    int k0 = t - 8; if (k0 < 0) k0 = 0;
    int k1 = t + 8; if (k1 > TT - 1) k1 = TT - 1;
    float sc[17];
    float mx = -1e30f;
    for (int k = k0; k <= k1; k++) {
        const float* kp = qkv + ((long)b * TT + k) * 768 + 256 + h * HD;
        float sacc = 0.0f;
        for (int d = 0; d < HD; d++) sacc += qr[d] * kp[d];
        sacc *= scale;
        sc[k - k0] = sacc;
        if (sacc > mx) mx = sacc;
    }
    float l = 0.0f, oa[HD];
    for (int d = 0; d < HD; d++) oa[d] = 0.0f;
    for (int k = k0; k <= k1; k++) {
        float p = expf(sc[k - k0] - mx);
        l += p;
        const float* vp = qkv + ((long)b * TT + k) * 768 + 512 + h * HD;
        for (int d = 0; d < HD; d++) oa[d] += p * vp[d];
    }
    float il = 1.0f / l;
    _Float16* op = o + ((long)b * TT + t) * HID + h * HD;
    for (int d = 0; d < HD; d++) op[d] = (_Float16)(oa[d] * il);
}

// Global attention, non-special rows (t%100!=0): only self key -> o = v.
__global__ void attn_glob_copy(const float* __restrict__ qkv, _Float16* __restrict__ o) {
    long i = blockIdx.x * (long)blockDim.x + threadIdx.x;
    long total = (long)BB * TT * HID;
    if (i >= total) return;
    int m = (int)(i % HID);
    long n = i / HID;
    int t = (int)(n % TT);
    if ((t % 100) == 0) return;
    o[i] = (_Float16)qkv[n * 768 + 512 + m];
}

// Global attention, special rows (q = 100*row): full softmax over T keys.
// One wave per (b,h,row); lanes split keys, two-pass, butterfly reductions.
__global__ void attn_glob_rows(const float* __restrict__ qkv, _Float16* __restrict__ o) {
    int lane = threadIdx.x;
    int q = blockIdx.x * 100;
    int h = blockIdx.y;
    int b = blockIdx.z;
    const float scale = 0.17677669529663687f;
    float qr[HD];
    const float* qp = qkv + ((long)b * TT + q) * 768 + h * HD;
    for (int d = 0; d < HD; d++) qr[d] = qp[d];
    float mx = -1e30f;
    for (int k = lane; k < TT; k += 32) {
        const float* kp = qkv + ((long)b * TT + k) * 768 + 256 + h * HD;
        float sacc = 0.0f;
        for (int d = 0; d < HD; d++) sacc += qr[d] * kp[d];
        sacc *= scale;
        if (sacc > mx) mx = sacc;
    }
    for (int off = 16; off >= 1; off >>= 1) {
        float o2 = __shfl_xor(mx, off, 32);
        if (o2 > mx) mx = o2;
    }
    float l = 0.0f, oa[HD];
    for (int d = 0; d < HD; d++) oa[d] = 0.0f;
    for (int k = lane; k < TT; k += 32) {
        const float* kp = qkv + ((long)b * TT + k) * 768 + 256 + h * HD;
        const float* vp = kp + 256;
        float sacc = 0.0f;
        for (int d = 0; d < HD; d++) sacc += qr[d] * kp[d];
        float p = expf(sacc * scale - mx);
        l += p;
        for (int d = 0; d < HD; d++) oa[d] += p * vp[d];
    }
    for (int off = 16; off >= 1; off >>= 1) l += __shfl_xor(l, off, 32);
    float myv = 0.0f;
    for (int d = 0; d < HD; d++) {
        float sv = oa[d];
        for (int off = 16; off >= 1; off >>= 1) sv += __shfl_xor(sv, off, 32);
        if (lane == d) myv = sv;
    }
    o[((long)b * TT + q) * HID + h * HD + lane] = (_Float16)(myv / l);
}

// ---------------------------------------------------------------------------
// Host orchestration
// ---------------------------------------------------------------------------
static inline long cdivl(long a, long b) { return (a + b - 1) / b; }

extern "C" void kernel_launch(void* const* d_in, const int* in_sizes, int n_in,
                              void* d_out, int out_size, void* d_ws, size_t ws_size,
                              hipStream_t stream) {
    (void)in_sizes; (void)n_in; (void)out_size; (void)ws_size;
    const float* X = (const float*)d_in[0];
    // conv params: base 1 + 6*(i-1): w, b, bn_scale, bn_bias, bn_mean, bn_var
    const float* CW[5]; const float* CBi[5]; const float* CS[5]; const float* CBB[5]; const float* CM[5]; const float* CV[5];
    for (int i = 1; i <= 4; i++) {
        int base = 1 + 6 * (i - 1);
        CW[i]  = (const float*)d_in[base + 0];
        CBi[i] = (const float*)d_in[base + 1];
        CS[i]  = (const float*)d_in[base + 2];
        CBB[i] = (const float*)d_in[base + 3];
        CM[i]  = (const float*)d_in[base + 4];
        CV[i]  = (const float*)d_in[base + 5];
    }
    const float* PROJW = (const float*)d_in[25];
    const float* PROJB = (const float*)d_in[26];

    char* ws = (char*)d_ws;
    const size_t MB = 1ull << 20;
    _Float16* W2P  = (_Float16*)(ws + 0 * MB);
    _Float16* W3P  = (_Float16*)(ws + 1 * MB);
    _Float16* W4P  = (_Float16*)(ws + 2 * MB);
    _Float16* WPRJ = (_Float16*)(ws + 4 * MB);
    _Float16* WQ[2]  = { (_Float16*)(ws + 7 * MB),  (_Float16*)(ws + 11 * MB) };
    _Float16* WO[2]  = { (_Float16*)(ws + 8 * MB),  (_Float16*)(ws + 12 * MB) };
    _Float16* WM1[2] = { (_Float16*)(ws + 9 * MB),  (_Float16*)(ws + 13 * MB) };
    _Float16* WM2[2] = { (_Float16*)(ws + 10 * MB), (_Float16*)(ws + 14 * MB) };
    _Float16* BUFA = (_Float16*)(ws + 16 * MB);    // h1 / h3 (dead before transformer)
    _Float16* BUFB = (_Float16*)(ws + 160 * MB);   // h2 / h4
    float*    R0   = (float*)(ws + 16 * MB);       // aliases BUFA region (safe)
    float*    R1   = (float*)(ws + 25 * MB);
    _Float16* XN   = (_Float16*)(ws + 34 * MB);
    float*    QKV  = (float*)(ws + 39 * MB);
    _Float16* OBUF = (_Float16*)(ws + 66 * MB);
    _Float16* MLPH = (_Float16*)(ws + 71 * MB);

    const int TPB = 256;
    const long NTOK = (long)BB * TT; // 8192

    // --- pack weights into WMMA A-layout (f32 -> f16) ---
    { long n = 25L * 4 * 1 * 512;  pack_conv_w<<<cdivl(n, TPB), TPB, 0, stream>>>(CW[2], W2P, 64, 32, 5, 5); }
    { long n = 49L * 8 * 2 * 512;  pack_conv_w<<<cdivl(n, TPB), TPB, 0, stream>>>(CW[3], W3P, 128, 64, 7, 7); }
    { long n = 49L * 8 * 4 * 512;  pack_conv_w<<<cdivl(n, TPB), TPB, 0, stream>>>(CW[4], W4P, 128, 128, 7, 7); }
    { long n = 256L * 4224;        pack_mat_w<<<cdivl(n, TPB), TPB, 0, stream>>>(PROJW, WPRJ, 256, 4224, 1); }
    for (int blk = 0; blk < 2; blk++) {
        int base = 27 + 12 * blk;
        { long n = 768L * 256;  pack_mat_w<<<cdivl(n, TPB), TPB, 0, stream>>>((const float*)d_in[base + 2], WQ[blk], 768, 256, 0); }
        { long n = 256L * 256;  pack_mat_w<<<cdivl(n, TPB), TPB, 0, stream>>>((const float*)d_in[base + 4], WO[blk], 256, 256, 0); }
        { long n = 1024L * 256; pack_mat_w<<<cdivl(n, TPB), TPB, 0, stream>>>((const float*)d_in[base + 8], WM1[blk], 1024, 256, 0); }
        { long n = 256L * 1024; pack_mat_w<<<cdivl(n, TPB), TPB, 0, stream>>>((const float*)d_in[base + 10], WM2[blk], 256, 1024, 0); }
    }

    // --- conv stack ---
    { long n = (long)BB * 257 * TT * 32;
      conv1_kernel<<<cdivl(n, TPB), TPB, 0, stream>>>(X, CW[1], CBi[1], CS[1], CBB[1], CM[1], CV[1], BUFA); }
    conv_wmma<<<dim3(TT / 16, 129, BB * 4), 32, 0, stream>>>(BUFA, W2P, CBi[2], CS[2], CBB[2], CM[2], CV[2], BUFB, 32, 64, 257, 129, 5, 2);
    conv_wmma<<<dim3(TT / 16, 65, BB * 8), 32, 0, stream>>>(BUFB, W3P, CBi[3], CS[3], CBB[3], CM[3], CV[3], BUFA, 64, 128, 129, 65, 7, 3);
    conv_wmma<<<dim3(TT / 16, 33, BB * 8), 32, 0, stream>>>(BUFA, W4P, CBi[4], CS[4], CBB[4], CM[4], CV[4], BUFB, 128, 128, 65, 33, 7, 3);

    // --- projection + positional encoding -> R0 (f32 residual stream) ---
    gemm_wmma<<<dim3(NTOK / 16, 256 / 16), 32, 0, stream>>>(WPRJ, BUFB, 256, 4224, GM_PROJ, PROJB, nullptr, R0, nullptr);

    // --- two transformer blocks ---
    for (int blk = 0; blk < 2; blk++) {
        int base = 27 + 12 * blk;
        const float* ln1s = (const float*)d_in[base + 0];
        const float* ln1b = (const float*)d_in[base + 1];
        const float* inb  = (const float*)d_in[base + 3];
        const float* outb = (const float*)d_in[base + 5];
        const float* ln2s = (const float*)d_in[base + 6];
        const float* ln2b = (const float*)d_in[base + 7];
        const float* m1b  = (const float*)d_in[base + 9];
        const float* m2b  = (const float*)d_in[base + 11];

        layernorm_k<<<NTOK, 32, 0, stream>>>(R0, ln1s, ln1b, XN);
        gemm_wmma<<<dim3(NTOK / 16, 768 / 16), 32, 0, stream>>>(WQ[blk], XN, 768, 256, GM_QKV, inb, nullptr, QKV, nullptr);
        if (blk == 0) {
            long n = (long)BB * HEADS * TT;
            attn_local<<<cdivl(n, TPB), TPB, 0, stream>>>(QKV, OBUF);
        } else {
            long n = (long)BB * TT * HID;
            attn_glob_copy<<<cdivl(n, TPB), TPB, 0, stream>>>(QKV, OBUF);
            attn_glob_rows<<<dim3(21, HEADS, BB), 32, 0, stream>>>(QKV, OBUF);
        }
        gemm_wmma<<<dim3(NTOK / 16, 256 / 16), 32, 0, stream>>>(WO[blk], OBUF, 256, 256, GM_RES, outb, R0, R1, nullptr);
        layernorm_k<<<NTOK, 32, 0, stream>>>(R1, ln2s, ln2b, XN);
        gemm_wmma<<<dim3(NTOK / 16, 1024 / 16), 32, 0, stream>>>(WM1[blk], XN, 1024, 256, GM_GELU, m1b, nullptr, nullptr, MLPH);
        float* dst = (blk == 1) ? (float*)d_out : R0;
        gemm_wmma<<<dim3(NTOK / 16, 256 / 16), 32, 0, stream>>>(WM2[blk], MLPH, 256, 1024, GM_RES, m2b, R1, dst, nullptr);
    }
}
